// EfficientAttention_42314017800858
// MI455X (gfx1250) — compile-verified, matrix operand order checked by reading, not run on hardware
//
#include <hip/hip_runtime.h>
#include <stdint.h>

typedef __bf16 bf16;
typedef __attribute__((ext_vector_type(16))) __bf16 v16bf;
typedef __attribute__((ext_vector_type(8)))  float  v8f;
typedef __attribute__((ext_vector_type(4)))  unsigned int u32x4;
typedef __attribute__((ext_vector_type(4)))  int v4i;

#define T_SEQ  2048
#define DMODEL 2048
#define NH     16
#define NKV    4
#define DH     128
#define KV_DIM 512
#define WINDOW 512

// ---- CDNA5 async global->LDS copy (ASYNCcnt path), with safe fallback ----
__device__ __forceinline__ void async_copy16(const bf16* g, bf16* l) {
#if __has_builtin(__builtin_amdgcn_global_load_async_to_lds_b128)
    __builtin_amdgcn_global_load_async_to_lds_b128(
        (__attribute__((address_space(1))) v4i*)(bf16*)g,
        (__attribute__((address_space(3))) v4i*)l, 0, 0);
#else
    *(u32x4*)l = *(const u32x4*)g;
#endif
}
__device__ __forceinline__ void async_wait0() {
#if __has_builtin(__builtin_amdgcn_s_wait_asynccnt)
    __builtin_amdgcn_s_wait_asynccnt(0);
#endif
}

// Load a 16x32 bf16 A-fragment (or B-fragment from an [N][K] LDS tile):
// lanes 0-15 take K = kb..kb+7 / kb+16..kb+23 with kb=0; lanes 16-31 kb=8.
__device__ __forceinline__ v16bf frag_ld(const bf16* p, int kb) {
    union U { v16bf v; u32x4 q[2]; };
    U t;
    t.q[0] = *(const u32x4*)(p + kb);
    t.q[1] = *(const u32x4*)(p + kb + 16);
    return t.v;
}

__global__ __launch_bounds__(256)
void f32_to_bf16(const float* __restrict__ src, bf16* __restrict__ dst, int n) {
    int i = blockIdx.x * blockDim.x + threadIdx.x;
    int stride = gridDim.x * blockDim.x;
    for (; i < n; i += stride) dst[i] = (bf16)src[i];
}

// C[M,N] = A[M,K] @ B[K,N], bf16 in, f32 out. 128x128 block, BK=32,
// double-buffered LDS with async global->LDS staging of the A panel.
// 8 waves arranged 4(m) x 2(n); each wave owns 32x64 = 2x4 WMMA tiles.
template<int N, int K>
__global__ __launch_bounds__(256)
void gemm_bf16_f32(const bf16* __restrict__ A, const bf16* __restrict__ B,
                   float* __restrict__ C)
{
    __shared__ __align__(16) bf16 As[2][128][40];   // [m][k], padded
    __shared__ __align__(16) bf16 Bs[2][128][40];   // [n][k], transposed, padded

    const int tid   = threadIdx.x;
    const int lane  = tid & 31;
    const int wave  = tid >> 5;
    const int m0    = blockIdx.y * 128;
    const int n0    = blockIdx.x * 128;
    const int wm    = (wave >> 1) * 32;
    const int wn    = (wave & 1) * 64;
    const int row16 = lane & 15;
    const int half  = lane >> 4;
    const int kb    = half * 8;

    auto stage = [&](int buf, int k0) {
        {   // A tile 128x32: thread -> row tid>>1, 16 elems at col (tid&1)*16
            int r = tid >> 1, c = (tid & 1) * 16;
            const bf16* src = A + (size_t)(m0 + r) * K + k0 + c;
            async_copy16(src,     &As[buf][r][c]);
            async_copy16(src + 8, &As[buf][r][c + 8]);
        }
        {   // B tile 32x128 -> Bs[n][k] (transpose on store)
            int kk = tid >> 3, c0 = (tid & 7) * 16;
            const bf16* src = B + (size_t)(k0 + kk) * N + n0 + c0;
#pragma unroll
            for (int i = 0; i < 16; ++i) Bs[buf][c0 + i][kk] = src[i];
        }
    };

    const v8f z8 = {0.f,0.f,0.f,0.f,0.f,0.f,0.f,0.f};
    v8f acc[2][4];
#pragma unroll
    for (int i = 0; i < 2; ++i)
#pragma unroll
        for (int j = 0; j < 4; ++j) acc[i][j] = z8;

    stage(0, 0);
    async_wait0();
    __syncthreads();

    for (int k0 = 0; k0 < K; k0 += 32) {
        const int cur = (k0 >> 5) & 1;
        if (k0 + 32 < K) stage(cur ^ 1, k0 + 32);   // overlap with WMMA below

        v16bf aF[2], bF[4];
#pragma unroll
        for (int mt = 0; mt < 2; ++mt) aF[mt] = frag_ld(&As[cur][wm + mt*16 + row16][0], kb);
#pragma unroll
        for (int nt = 0; nt < 4; ++nt) bF[nt] = frag_ld(&Bs[cur][wn + nt*16 + row16][0], kb);
#pragma unroll
        for (int mt = 0; mt < 2; ++mt)
#pragma unroll
            for (int nt = 0; nt < 4; ++nt)
                acc[mt][nt] = __builtin_amdgcn_wmma_f32_16x16x32_bf16(
                    false, aF[mt], false, bF[nt], (short)0, acc[mt][nt], false, false);

        async_wait0();
        __syncthreads();
    }

    // store: N is constexpr -> row/col offsets fold into instruction IOFFSETs
#pragma unroll
    for (int mt = 0; mt < 2; ++mt) {
        float* cb = C + (size_t)(m0 + wm + mt*16 + 8*half) * N + (n0 + wn + row16);
#pragma unroll
        for (int nt = 0; nt < 4; ++nt)
#pragma unroll
            for (int r = 0; r < 8; ++r)
                cb[r*N + nt*16] = acc[mt][nt][r];
    }
}

// RoPE on q/k, convert v; repack to head-major bf16 [head][T][128].
__global__ __launch_bounds__(64)
void rope_pack(const float* __restrict__ qf, const float* __restrict__ kf,
               const float* __restrict__ vf, const float* __restrict__ cosT,
               const float* __restrict__ sinT, bf16* __restrict__ qpk,
               bf16* __restrict__ kpk, bf16* __restrict__ vpk)
{
    const int t = blockIdx.x, h = blockIdx.y, d = threadIdx.x;  // d in [0,64)
    const float c1 = cosT[t*DH + d],      s1 = sinT[t*DH + d];
    const float c2 = cosT[t*DH + d + 64], s2 = sinT[t*DH + d + 64];

    const float* q = qf + (size_t)t*DMODEL + h*DH;
    bf16* qo = qpk + ((size_t)h*T_SEQ + t)*DH;
    float q1 = q[d], q2 = q[d + 64];
    qo[d]      = (bf16)(q1*c1 - q2*s1);
    qo[d + 64] = (bf16)(q2*c2 + q1*s2);

    if (h < NKV) {
        const float* k = kf + (size_t)t*KV_DIM + h*DH;
        bf16* ko = kpk + ((size_t)h*T_SEQ + t)*DH;
        float k1 = k[d], k2 = k[d + 64];
        ko[d]      = (bf16)(k1*c1 - k2*s1);
        ko[d + 64] = (bf16)(k2*c2 + k1*s2);
        const float* v = vf + (size_t)t*KV_DIM + h*DH;
        bf16* vo = vpk + ((size_t)h*T_SEQ + t)*DH;
        vo[d]      = (bf16)v[d];
        vo[d + 64] = (bf16)v[d + 64];
    }
}

// Flash-style sliding-window attention. Block = (128 q rows, 1 head), 8 waves,
// each wave owns 16 q rows. KV streamed in tiles of 64 keys.
__global__ __launch_bounds__(256)
void attn_swa(const bf16* __restrict__ qpk, const bf16* __restrict__ kpk,
              const bf16* __restrict__ vpk, bf16* __restrict__ yb)
{
    __shared__ __align__(16) bf16 ks [64][136];   // [key][d]
    __shared__ __align__(16) bf16 vts[128][72];   // [d][key] (transposed)
    __shared__ __align__(16) bf16 ps [8][16][72]; // per-wave P (row-major)

    const int qs0   = blockIdx.x * 128;
    const int h     = blockIdx.y;
    const int kvh   = h >> 2;
    const int tid   = threadIdx.x;
    const int lane  = tid & 31;
    const int wave  = tid >> 5;
    const int row16 = lane & 15;
    const int half  = lane >> 4;
    const int kb    = half * 8;
    const float scale = 0.08838834764831845f;  // 1/sqrt(128)

    // Q fragments for this wave's 16 rows, 4 K-chunks of 32 (held in VGPRs)
    v16bf qF[4];
    {
        const bf16* qrow = qpk + ((size_t)h*T_SEQ + qs0 + wave*16 + row16) * DH;
#pragma unroll
        for (int c = 0; c < 4; ++c) qF[c] = frag_ld(qrow + c*32, kb);
    }

    const v8f z8 = {0.f,0.f,0.f,0.f,0.f,0.f,0.f,0.f};
    v8f accO[8];
#pragma unroll
    for (int i = 0; i < 8; ++i) accO[i] = z8;
    float mOld[8], lSum[8];
#pragma unroll
    for (int r = 0; r < 8; ++r) { mOld[r] = -3.0e38f; lSum[r] = 0.f; }

    const int jt0 = (qs0 >= WINDOW) ? (qs0 - WINDOW) : 0;
    for (int jt = jt0; jt < qs0 + 128; jt += 64) {
        __syncthreads();
        {   // stage K tile [64][128] async; V tile transposed [128][64] manual
            int key = tid >> 2, dseg = (tid & 3) * 32;
            const bf16* gk = kpk + ((size_t)kvh*T_SEQ + jt + key)*DH + dseg;
            bf16* lk = &ks[key][dseg];
#pragma unroll
            for (int i = 0; i < 4; ++i) async_copy16(gk + i*8, lk + i*8);
            const bf16* gv = vpk + ((size_t)kvh*T_SEQ + jt + key)*DH + dseg;
#pragma unroll
            for (int i = 0; i < 32; ++i) vts[dseg + i][key] = gv[i];
        }
        async_wait0();
        __syncthreads();

        // S = Q K^T : 4 key-tiles x 4 d-chunks of 32
        v8f accS[4];
#pragma unroll
        for (int i = 0; i < 4; ++i) accS[i] = z8;
#pragma unroll
        for (int jti = 0; jti < 4; ++jti)
#pragma unroll
            for (int dc = 0; dc < 4; ++dc) {
                v16bf bK = frag_ld(&ks[jti*16 + row16][dc*32], kb);
                accS[jti] = __builtin_amdgcn_wmma_f32_16x16x32_bf16(
                    false, qF[dc], false, bK, (short)0, accS[jti], false, false);
            }

        // scale + sliding-window mask + row max (per-lane over 4 tiles)
        float tmax[8];
#pragma unroll
        for (int r = 0; r < 8; ++r) tmax[r] = -3.0e38f;
#pragma unroll
        for (int jti = 0; jti < 4; ++jti)
#pragma unroll
            for (int r = 0; r < 8; ++r) {
                int i = qs0 + wave*16 + r + 8*half;
                int j = jt + jti*16 + row16;
                float s = accS[jti][r] * scale;
                bool ok = (j <= i) && (j > i - WINDOW);
                s = ok ? s : -1.0e30f;
                accS[jti][r] = s;
                tmax[r] = fmaxf(tmax[r], s);
            }

        float mNew[8], corr[8];
#pragma unroll
        for (int r = 0; r < 8; ++r) {
            float v = tmax[r];
            v = fmaxf(v, __shfl_xor(v, 1, 32));
            v = fmaxf(v, __shfl_xor(v, 2, 32));
            v = fmaxf(v, __shfl_xor(v, 4, 32));
            v = fmaxf(v, __shfl_xor(v, 8, 32));   // reduce across 16-lane half
            mNew[r] = fmaxf(mOld[r], v);
            corr[r] = __expf(mOld[r] - mNew[r]);
            mOld[r] = mNew[r];
        }

        // P = exp(S - m); write to per-wave LDS (C-layout -> row-major)
        float psum[8];
#pragma unroll
        for (int r = 0; r < 8; ++r) psum[r] = 0.f;
#pragma unroll
        for (int jti = 0; jti < 4; ++jti)
#pragma unroll
            for (int r = 0; r < 8; ++r) {
                float p = __expf(accS[jti][r] - mNew[r]);
                psum[r] += p;
                ps[wave][r + 8*half][jti*16 + row16] = (bf16)p;
            }
#pragma unroll
        for (int r = 0; r < 8; ++r) {
            float v = psum[r];
            v += __shfl_xor(v, 1, 32);
            v += __shfl_xor(v, 2, 32);
            v += __shfl_xor(v, 4, 32);
            v += __shfl_xor(v, 8, 32);
            lSum[r] = lSum[r]*corr[r] + v;
        }

        // rescale O, then O += P V
#pragma unroll
        for (int dt = 0; dt < 8; ++dt)
#pragma unroll
            for (int r = 0; r < 8; ++r) accO[dt][r] *= corr[r];
#pragma unroll
        for (int kc = 0; kc < 2; ++kc) {
            v16bf aP = frag_ld(&ps[wave][row16][kc*32], kb);   // intra-wave LDS RAW: in-order
#pragma unroll
            for (int dt = 0; dt < 8; ++dt) {
                v16bf bV = frag_ld(&vts[dt*16 + row16][kc*32], kb);
                accO[dt] = __builtin_amdgcn_wmma_f32_16x16x32_bf16(
                    false, aP, false, bV, (short)0, accO[dt], false, false);
            }
        }
    }

    // normalize and write y as [t][h*128+d] bf16 for the final GEMM
#pragma unroll
    for (int dt = 0; dt < 8; ++dt)
#pragma unroll
        for (int r = 0; r < 8; ++r) {
            int t = qs0 + wave*16 + r + 8*half;
            yb[(size_t)t*DMODEL + h*DH + dt*16 + row16] =
                (bf16)(accO[dt][r] / lSum[r]);
        }
}

extern "C" void kernel_launch(void* const* d_in, const int* in_sizes, int n_in,
                              void* d_out, int out_size, void* d_ws, size_t ws_size,
                              hipStream_t stream)
{
    (void)in_sizes; (void)n_in; (void)out_size; (void)ws_size;
    const float* x    = (const float*)d_in[0];
    const float* cosT = (const float*)d_in[1];
    const float* sinT = (const float*)d_in[2];
    const float* Wq   = (const float*)d_in[3];
    const float* Wk   = (const float*)d_in[4];
    const float* Wv   = (const float*)d_in[5];
    const float* Wo   = (const float*)d_in[6];
    float* out = (float*)d_out;

    char* ws = (char*)d_ws;
    size_t off = 0;
    auto alloc = [&](size_t bytes) {
        char* p = ws + off;
        off = (off + bytes + 255) & ~(size_t)255;
        return (void*)p;
    };
    bf16*  xb  = (bf16*) alloc((size_t)T_SEQ*DMODEL*2);
    bf16*  wqb = (bf16*) alloc((size_t)DMODEL*DMODEL*2);
    bf16*  wkb = (bf16*) alloc((size_t)DMODEL*KV_DIM*2);
    bf16*  wvb = (bf16*) alloc((size_t)DMODEL*KV_DIM*2);
    bf16*  wob = (bf16*) alloc((size_t)DMODEL*DMODEL*2);
    float* qf  = (float*)alloc((size_t)T_SEQ*DMODEL*4);
    float* kf  = (float*)alloc((size_t)T_SEQ*KV_DIM*4);
    float* vf  = (float*)alloc((size_t)T_SEQ*KV_DIM*4);
    bf16*  qpk = (bf16*) alloc((size_t)NH *T_SEQ*DH*2);
    bf16*  kpk = (bf16*) alloc((size_t)NKV*T_SEQ*DH*2);
    bf16*  vpk = (bf16*) alloc((size_t)NKV*T_SEQ*DH*2);
    bf16*  yb  = (bf16*) alloc((size_t)T_SEQ*DMODEL*2);

    auto cvt = [&](const float* s, bf16* d, int n) {
        int blocks = (n + 1023) / 1024;
        f32_to_bf16<<<blocks, 256, 0, stream>>>(s, d, n);
    };
    cvt(x,  xb,  T_SEQ*DMODEL);
    cvt(Wq, wqb, DMODEL*DMODEL);
    cvt(Wk, wkb, DMODEL*KV_DIM);
    cvt(Wv, wvb, DMODEL*KV_DIM);
    cvt(Wo, wob, DMODEL*DMODEL);

    gemm_bf16_f32<DMODEL, DMODEL><<<dim3(DMODEL/128, T_SEQ/128), 256, 0, stream>>>(xb, wqb, qf);
    gemm_bf16_f32<KV_DIM, DMODEL><<<dim3(KV_DIM/128, T_SEQ/128), 256, 0, stream>>>(xb, wkb, kf);
    gemm_bf16_f32<KV_DIM, DMODEL><<<dim3(KV_DIM/128, T_SEQ/128), 256, 0, stream>>>(xb, wvb, vf);

    rope_pack<<<dim3(T_SEQ, NH), 64, 0, stream>>>(qf, kf, vf, cosT, sinT, qpk, kpk, vpk);

    attn_swa<<<dim3(T_SEQ/128, NH), 256, 0, stream>>>(qpk, kpk, vpk, yb);

    gemm_bf16_f32<DMODEL, DMODEL><<<dim3(DMODEL/128, T_SEQ/128), 256, 0, stream>>>(yb, wob, out);
}
